// SphericalContraction_77876347011738
// MI455X (gfx1250) — matrix operand order repaired and checked by reading, not compiled, not run on hardware
//
#include <hip/hip_runtime.h>
#include <hip/hip_bf16.h>

// ---------------------------------------------------------------------------
// Problem constants (from the reference)
// ---------------------------------------------------------------------------
#define N_TYPES     4
#define N_SAMPLES   512
#define N_RADIAL    64
#define OUT_FEAT    256
#define IN_FEAT     16384              // 4 * 64 * 64
#define M_TOTAL     (N_SAMPLES * 3)    // 1536   (s*3 + m)
#define K_TOTAL     IN_FEAT
#define N_TOTAL     OUT_FEAT

typedef __attribute__((ext_vector_type(16))) __bf16        v16bf;
typedef __attribute__((ext_vector_type(8)))  float         v8f;
typedef __attribute__((ext_vector_type(4)))  unsigned int  v4u;
typedef __attribute__((ext_vector_type(8)))  int           v8i;
typedef __attribute__((ext_vector_type(4)))  int           v4i;

static __device__ __forceinline__ __bf16 to_bf16(float f) { return (__bf16)f; }

// Generic LDS pointer -> LDS byte offset (ISA: LDS aperture addr[31:0] is the
// LDS address; upper bits are the shared-base aperture tag).
static __device__ __forceinline__ unsigned lds_off(const void* p) {
    return (unsigned)(uintptr_t)p;
}

// ---------------------------------------------------------------------------
// Kernel 1: CG feature construction  ->  feats[z][M][K] in bf16
//   feat[m,p,P] = sum_{c,C} cg[c,C,m] * x1[c,p] * x2[C,P]
// Grid: (s=512, z=4, l1=4), 256 threads.
// ---------------------------------------------------------------------------
__global__ __launch_bounds__(256)
void feat_kernel(const float* __restrict__ x0, const float* __restrict__ x1,
                 const float* __restrict__ x2, const float* __restrict__ x3,
                 const float* __restrict__ x4,
                 const float* __restrict__ cg0, const float* __restrict__ cg1,
                 const float* __restrict__ cg2, const float* __restrict__ cg3,
                 __bf16* __restrict__ feats)
{
    const int s  = blockIdx.x;
    const int z  = blockIdx.y;
    const int l1 = blockIdx.z;
    const int n1 = 2 * l1 + 1;   // <= 7
    const int n2 = 2 * l1 + 3;   // <= 9

    const float* xs[5]  = {x0, x1, x2, x3, x4};
    const float* cgs[4] = {cg0, cg1, cg2, cg3};
    const float* X1 = xs[l1];
    const float* X2 = xs[l1 + 1];
    const float* CG = cgs[l1];

    __shared__ float x1s[7 * 64];
    __shared__ float x2s[9 * 64];
    __shared__ float cgsh[7 * 9 * 3];

    const int tid = threadIdx.x;
    const size_t b1 = (size_t)(z * N_SAMPLES + s) * n1 * 64;
    const size_t b2 = (size_t)(z * N_SAMPLES + s) * n2 * 64;
    for (int i = tid; i < n1 * 64; i += 256)     x1s[i]  = X1[b1 + i];
    for (int i = tid; i < n2 * 64; i += 256)     x2s[i]  = X2[b2 + i];
    for (int i = tid; i < n1 * n2 * 3; i += 256) cgsh[i] = CG[i];
    __syncthreads();

    const int p  = tid >> 2;          // 0..63
    const int q  = tid & 3;           // P-chunk 0..3
    const int P0 = q * 16;

    float a[7];
#pragma unroll
    for (int c = 0; c < 7; ++c) a[c] = (c < n1) ? x1s[c * 64 + p] : 0.0f;

    float u[3][9];
#pragma unroll
    for (int m = 0; m < 3; ++m) {
#pragma unroll
        for (int C = 0; C < 9; ++C) {
            float sAcc = 0.0f;
#pragma unroll
            for (int c = 0; c < 7; ++c) {
                if (c < n1 && C < n2) sAcc += cgsh[(c * n2 + C) * 3 + m] * a[c];
            }
            u[m][C] = sAcc;
        }
    }

    const size_t rowBase = (size_t)(z * M_TOTAL + s * 3);
    const int    kBase   = l1 * 4096 + p * 64 + P0;

#pragma unroll
    for (int m = 0; m < 3; ++m) {
        __align__(16) __bf16 vals[16];
#pragma unroll
        for (int pp = 0; pp < 16; ++pp) {
            float acc = 0.0f;
#pragma unroll
            for (int C = 0; C < 9; ++C) {
                if (C < n2) acc += u[m][C] * x2s[C * 64 + P0 + pp];
            }
            vals[pp] = to_bf16(acc);
        }
        __bf16* dst = feats + (rowBase + m) * K_TOTAL + kBase;
        ((uint4*)dst)[0] = ((const uint4*)vals)[0];
        ((uint4*)dst)[1] = ((const uint4*)vals)[1];
    }
}

// ---------------------------------------------------------------------------
// Kernel 2: W[z][K][N] f32  ->  Wt[z][N][K] bf16  (LDS tile transpose)
// ---------------------------------------------------------------------------
__global__ __launch_bounds__(256)
void wconv_kernel(const float* __restrict__ W, __bf16* __restrict__ Wt)
{
    __shared__ float tile[32][33];
    const int z  = blockIdx.z;
    const int f0 = blockIdx.x * 32;
    const int o0 = blockIdx.y * 32;
    const int tx = threadIdx.x, ty = threadIdx.y;

    const float* Wz = W + (size_t)z * K_TOTAL * N_TOTAL;
#pragma unroll
    for (int i = 0; i < 32; i += 8)
        tile[ty + i][tx] = Wz[(size_t)(f0 + ty + i) * N_TOTAL + o0 + tx];
    __syncthreads();

    __bf16* Bz = Wt + (size_t)z * N_TOTAL * K_TOTAL;
#pragma unroll
    for (int i = 0; i < 32; i += 8)
        Bz[(size_t)(o0 + ty + i) * K_TOTAL + f0 + tx] = to_bf16(tile[tx][ty + i]);
}

// ---------------------------------------------------------------------------
// TDM descriptor helpers (CDNA5 ISA)
// ---------------------------------------------------------------------------
#define GEMM_BM 128
#define GEMM_BN 64
#define GEMM_BK 64
#define LDS_LDA (GEMM_BK + 16)   // 80 bf16 = 160B row stride (TDM pad: +8 DW / 32 DW)
#define LDS_LDB (GEMM_BK + 16)

static __device__ __forceinline__ v4u tdm_group0(unsigned ldsAddr, unsigned long long gaddr)
{
    v4u g;
    g[0] = 1u;                                           // count=1, user-mode, no gather
    g[1] = ldsAddr;                                      // lds_addr [63:32]
    g[2] = (unsigned)(gaddr & 0xFFFFFFFFu);              // global_addr[31:0]
    g[3] = (unsigned)((gaddr >> 32) & 0x01FFFFFFu)       // global_addr[56:32]
         | (2u << 30);                                   // type=2 ("image")
    return g;
}

static __device__ __forceinline__ v8i tdm_group1(unsigned d0, unsigned d1,
                                                 unsigned t0, unsigned t1,
                                                 unsigned long long s0 /*elems*/)
{
    v8i g;
    // data_size=1 (2B), pad_enable=1, pad_interval=4 (32 DW = 128B),
    // pad_amount=7 (8 DW = 32B) -> LDS row stride 160B = 80 bf16
    g[0] = (int)((1u << 16) | (1u << 20) | (4u << 22) | (7u << 25));
    g[1] = (int)((d0 & 0xFFFFu) << 16);                          // dim0[15:0] @ [63:48]
    g[2] = (int)(((d0 >> 16) & 0xFFFFu) | ((d1 & 0xFFFFu) << 16));
    g[3] = (int)(((d1 >> 16) & 0xFFFFu) | ((t0 & 0xFFFFu) << 16));
    g[4] = (int)(t1 & 0xFFFFu);                                  // tile_dim1; tile_dim2=0
    g[5] = (int)(s0 & 0xFFFFFFFFull);                            // dim0_stride[31:0]
    g[6] = (int)((s0 >> 32) & 0xFFFFull);                        // dim0_stride[47:32]
    g[7] = 0;
    return g;
}

static __device__ __forceinline__ void tdm_load_tile(unsigned ldsAddr,
                                                     const __bf16* gptr,
                                                     unsigned rows)
{
    const v4u g0 = tdm_group0(ldsAddr, (unsigned long long)(uintptr_t)gptr);
    const v8i g1 = tdm_group1(/*tensor_d0=*/K_TOTAL, /*tensor_d1=*/rows,
                              /*tile_d0=*/GEMM_BK,  /*tile_d1=*/rows,
                              /*d0_stride=*/(unsigned long long)K_TOTAL);
    const v4i z4 = {0, 0, 0, 0};
#if __clang_major__ >= 23
    const v8i z8 = {0, 0, 0, 0, 0, 0, 0, 0};
    __builtin_amdgcn_tensor_load_to_lds(g0, g1, z4, z4, z8, 0);
#else
    __builtin_amdgcn_tensor_load_to_lds(g0, g1, z4, z4, 0);
#endif
}

// ---------------------------------------------------------------------------
// Kernel 3: per-type GEMM, V_WMMA_F32_16X16X32_BF16 with TDM double-buffering
//   A = feats[z]  (1536 x 16384 bf16 row-major)
//   B = Wt[z]     (256  x 16384 bf16 row-major == W^T)
//   C = out[z]    (1536 x 256 f32 row-major)  == (z,s,3,256) flat
// ---------------------------------------------------------------------------
__global__ __launch_bounds__(256)
void gemm_kernel(const __bf16* __restrict__ A, const __bf16* __restrict__ B,
                 float* __restrict__ C)
{
    const int z = blockIdx.z;
    const __bf16* Az = A + (size_t)z * M_TOTAL * K_TOTAL;
    const __bf16* Bz = B + (size_t)z * N_TOTAL * K_TOTAL;
    float*        Cz = C + (size_t)z * M_TOTAL * N_TOTAL;

    const int m_blk = blockIdx.x * GEMM_BM;
    const int n_blk = blockIdx.y * GEMM_BN;

    __shared__ __bf16 As[2][GEMM_BM * LDS_LDA];   // 2 x 20 KB
    __shared__ __bf16 Bs[2][GEMM_BN * LDS_LDB];   // 2 x 10 KB

    const int tid  = threadIdx.x;
    const int lane = tid & 31;
    const int wave = tid >> 5;
    const int wm   = wave >> 1;   // 0..3 -> 32-row slab
    const int wn   = wave & 1;    // 0..1 -> 32-col slab
    const int lr   = lane & 15;
    const int lh   = lane >> 4;   // lane half

    const v8f vzero = {0.f, 0.f, 0.f, 0.f, 0.f, 0.f, 0.f, 0.f};
    v8f acc[2][2];
#pragma unroll
    for (int i = 0; i < 2; ++i)
#pragma unroll
        for (int j = 0; j < 2; ++j) acc[i][j] = vzero;

    constexpr int NT = K_TOTAL / GEMM_BK;   // 256 K-tiles

    // Prologue: DMA tile 0 into buffer 0 (TDM ignores EXEC; issue from wave 0
    // only via a wave-uniform branch so the copy happens exactly once).
    if (wave == 0) {
        tdm_load_tile(lds_off(&As[0][0]), Az + (size_t)m_blk * K_TOTAL, GEMM_BM);
        tdm_load_tile(lds_off(&Bs[0][0]), Bz + (size_t)n_blk * K_TOTAL, GEMM_BN);
    }

    int buf = 0;
    for (int t = 0; t < NT; ++t) {
        const int  k0      = t * GEMM_BK;
        const bool hasNext = (t + 1 < NT);

        // Prefetch next K-tile into the other buffer while we compute.
        if (wave == 0 && hasNext) {
            tdm_load_tile(lds_off(&As[buf ^ 1][0]),
                          Az + (size_t)m_blk * K_TOTAL + (k0 + GEMM_BK), GEMM_BM);
            tdm_load_tile(lds_off(&Bs[buf ^ 1][0]),
                          Bz + (size_t)n_blk * K_TOTAL + (k0 + GEMM_BK), GEMM_BN);
        }
        // Wait for current tile's pair (only the 2 prefetch ops may remain).
        if (hasNext) __builtin_amdgcn_s_wait_tensorcnt(2);
        else         __builtin_amdgcn_s_wait_tensorcnt(0);
        __syncthreads();

        const __bf16* Asb = &As[buf][0];
        const __bf16* Bsb = &Bs[buf][0];

#pragma unroll
        for (int kk = 0; kk < GEMM_BK; kk += 32) {
            // A frags (ISA 16-bit A 16x32: lane-half K {0..7,16..23}/{8..15,24..31})
            v16bf af[2];
#pragma unroll
            for (int mi = 0; mi < 2; ++mi) {
                const int row = wm * 32 + mi * 16 + lr;
                const __bf16* p = Asb + row * LDS_LDA + kk + lh * 8;
                v16bf f;
                ((uint4*)&f)[0] = *(const uint4*)p;
                ((uint4*)&f)[1] = *(const uint4*)(p + 16);
                af[mi] = f;
            }
            // B frags (lane-half K {0..7,16..23}/{8..15,24..31}, N = lane&15)
            v16bf bfm[2];
#pragma unroll
            for (int ni = 0; ni < 2; ++ni) {
                const int nrow = wn * 32 + ni * 16 + lr;
                const __bf16* p = Bsb + nrow * LDS_LDB + kk + lh * 8;
                v16bf f;
                ((uint4*)&f)[0] = *(const uint4*)p;
                ((uint4*)&f)[1] = *(const uint4*)(p + 16);
                bfm[ni] = f;
            }
#pragma unroll
            for (int mi = 0; mi < 2; ++mi)
#pragma unroll
                for (int ni = 0; ni < 2; ++ni)
                    acc[mi][ni] = __builtin_amdgcn_wmma_f32_16x16x32_bf16(
                        false, af[mi], false, bfm[ni],
                        (short)0, acc[mi][ni], false, false);
        }
        __syncthreads();   // all reads of buf done before TDM refills it (t+2)
        buf ^= 1;
    }

    // Epilogue: 16x16 f32 D layout (VGPR v -> row v / v+8 per lane half)
#pragma unroll
    for (int mi = 0; mi < 2; ++mi) {
#pragma unroll
        for (int ni = 0; ni < 2; ++ni) {
            const int mb = m_blk + wm * 32 + mi * 16 + lh * 8;
            const int nn = n_blk + wn * 32 + ni * 16 + lr;
#pragma unroll
            for (int v = 0; v < 8; ++v)
                Cz[(size_t)(mb + v) * N_TOTAL + nn] = acc[mi][ni][v];
        }
    }
}

// ---------------------------------------------------------------------------
// Launcher
// ---------------------------------------------------------------------------
extern "C" void kernel_launch(void* const* d_in, const int* in_sizes, int n_in,
                              void* d_out, int out_size, void* d_ws, size_t ws_size,
                              hipStream_t stream)
{
    const float* x0  = (const float*)d_in[0];
    const float* x1  = (const float*)d_in[1];
    const float* x2  = (const float*)d_in[2];
    const float* x3  = (const float*)d_in[3];
    const float* x4  = (const float*)d_in[4];
    const float* W   = (const float*)d_in[5];
    const float* cg0 = (const float*)d_in[6];
    const float* cg1 = (const float*)d_in[7];
    const float* cg2 = (const float*)d_in[8];
    const float* cg3 = (const float*)d_in[9];

    const size_t featsBytes = (size_t)N_TYPES * M_TOTAL * K_TOTAL * sizeof(__bf16); // ~201 MB
    __bf16* feats = (__bf16*)d_ws;
    __bf16* Wt    = (__bf16*)((char*)d_ws + featsBytes);                            // ~33.5 MB
    float*  out   = (float*)d_out;

    feat_kernel<<<dim3(N_SAMPLES, N_TYPES, 4), 256, 0, stream>>>(
        x0, x1, x2, x3, x4, cg0, cg1, cg2, cg3, feats);

    wconv_kernel<<<dim3(K_TOTAL / 32, N_TOTAL / 32, N_TYPES), dim3(32, 8), 0, stream>>>(
        W, Wt);

    gemm_kernel<<<dim3(M_TOTAL / GEMM_BM, N_TOTAL / GEMM_BN, N_TYPES), 256, 0, stream>>>(
        feats, Wt, out);
}
